// MultiHeadAttention_24412594111142
// MI455X (gfx1250) — matrix operand order repaired and checked by reading, not compiled, not run on hardware
//
#include <hip/hip_runtime.h>

// ---------------------------------------------------------------------------
// MHA forward for MI455X (gfx1250, wave32, WMMA).
// Roofline: ~69 GFLOP vs ~385 MB mandatory HBM traffic (attn_bias = 268 MB
// dominates) -> memory bound (~16.5us floor @ 23.3 TB/s). So: bf16 WMMA for
// all matmuls (f32 accumulate), non-temporal streaming of bias/mask, K/V tiles
// async-copied to LDS for 4x intra-WG reuse, flash-style online softmax so
// the (B,H,L,L) score tensor never touches memory.
// ---------------------------------------------------------------------------

#define B_  2
#define L_  2048
#define D_  1024
#define H_  16
#define DT_ 64

typedef __attribute__((ext_vector_type(16))) __bf16 v16bf;
typedef __attribute__((ext_vector_type(8)))  float  v8f;

#if defined(__has_builtin)
#if __has_builtin(__builtin_amdgcn_global_load_async_to_lds_b128)
#define USE_ASYNC_LDS 1
#endif
#endif

// async-copy builtin takes pointers to int4 vectors in AS(1)/AS(3)
typedef int vint4_t __attribute__((__vector_size__(4 * sizeof(int))));
typedef __attribute__((address_space(1))) vint4_t* gv4_p;  // global
typedef __attribute__((address_space(3))) vint4_t* lv4_p;  // LDS

__device__ __forceinline__ unsigned short f2bf_bits(float f) {
  unsigned u = __builtin_bit_cast(unsigned, f);
  u += 0x7FFFu + ((u >> 16) & 1u);           // round-to-nearest-even
  return (unsigned short)(u >> 16);
}
__device__ __forceinline__ __bf16 f2bf(float f) {
  unsigned short s = f2bf_bits(f);
  return __builtin_bit_cast(__bf16, s);
}
__device__ __forceinline__ v8f wmma_bf16(v16bf a, v16bf b, v8f c) {
  // D(16x16,f32) = A(16x32,bf16) * B(32x16,bf16) + C
  return __builtin_amdgcn_wmma_f32_16x16x32_bf16(false, a, false, b,
                                                 (short)0, c, false, false);
}

// ---------------------------------------------------------------------------
// Projection GEMM: out = X @ W^T + bias, X:(B*L, D) f32, W:(D, D) f32.
// WG = 256 thr (8 waves), tile 64(M) x 64(N); wave = 16(M) x 32(N).
// VMODE 0: store bf16 as (B,H,L,DT)   (used for q and k)
// VMODE 1: store bf16 as (B,H,DT,L)   (v transposed -> p@v B-operand is
//                                      contiguous per lane later)
// A-frag per lane:  a[8g+j]  = X[m][k0 + half*16 + g*8 + j]   (contiguous 16)
// B-frag per lane:  b[j]     = W[n][k0 + half*16 + j]         (contiguous 16)
// ---------------------------------------------------------------------------
template <int VMODE>
__global__ __launch_bounds__(256) void proj_kernel(
    const float* __restrict__ X, const float* __restrict__ W,
    const float* __restrict__ bias, unsigned short* __restrict__ out) {
  const int tid  = threadIdx.x;
  const int lane = tid & 31, wid = tid >> 5;
  const int l16  = lane & 15, lh = lane >> 4;
  const int m0 = blockIdx.y * 64 + (wid & 3) * 16;
  const int n0 = blockIdx.x * 64 + (wid >> 2) * 32;
  const int m_a = m0 + l16;

  v8f c0 = {0.f,0.f,0.f,0.f,0.f,0.f,0.f,0.f}, c1 = c0;
  const float* pa  = X + (size_t)m_a * D_ + lh * 16;
  const float* pb0 = W + (size_t)(n0 + l16) * D_ + lh * 16;
  const float* pb1 = pb0 + (size_t)16 * D_;

  for (int k0 = 0; k0 < D_; k0 += 32) {
    v16bf a, b0, b1;
#pragma unroll
    for (int g = 0; g < 2; ++g) {
      float4 f0 = *(const float4*)(pa + k0 + g * 8);
      float4 f1 = *(const float4*)(pa + k0 + g * 8 + 4);
      a[8*g+0]=f2bf(f0.x); a[8*g+1]=f2bf(f0.y); a[8*g+2]=f2bf(f0.z); a[8*g+3]=f2bf(f0.w);
      a[8*g+4]=f2bf(f1.x); a[8*g+5]=f2bf(f1.y); a[8*g+6]=f2bf(f1.z); a[8*g+7]=f2bf(f1.w);
    }
#pragma unroll
    for (int q = 0; q < 4; ++q) {
      float4 f0 = *(const float4*)(pb0 + k0 + q * 4);
      float4 f1 = *(const float4*)(pb1 + k0 + q * 4);
      b0[4*q+0]=f2bf(f0.x); b0[4*q+1]=f2bf(f0.y); b0[4*q+2]=f2bf(f0.z); b0[4*q+3]=f2bf(f0.w);
      b1[4*q+0]=f2bf(f1.x); b1[4*q+1]=f2bf(f1.y); b1[4*q+2]=f2bf(f1.z); b1[4*q+3]=f2bf(f1.w);
    }
    c0 = wmma_bf16(a, b0, c0);
    c1 = wmma_bf16(a, b1, c1);
  }

#pragma unroll
  for (int r = 0; r < 8; ++r) {
    int m = m0 + r + 8 * lh;             // C-layout row
    int bb = m / L_, l = m % L_;
#pragma unroll
    for (int t = 0; t < 2; ++t) {
      int n = n0 + t * 16 + l16;         // C-layout col
      float v = ((t == 0) ? c0[r] : c1[r]) + bias[n];
      int h = n >> 6, dt = n & 63;
      size_t idx;
      if (VMODE) idx = (((size_t)bb * H_ + h) * DT_ + dt) * L_ + l;   // v^T
      else       idx = (((size_t)bb * H_ + h) * L_ + l) * DT_ + dt;   // q,k
      out[idx] = f2bf_bits(v);
    }
  }
}

// ---------------------------------------------------------------------------
// Flash attention: WG = 128 thr (4 waves) handles 64 q rows of one (b,h).
// Loop over 32 key-blocks of 64: stage K (64x64, [key][d]) and V^T
// (64x64, [dt][key]) bf16 tiles into LDS via async copies (4x reuse across
// waves), scores via 8 WMMAs, mask/bias via non-temporal loads, online
// softmax (shfl reductions within 16-lane C-layout row groups), P transposed
// D-layout -> A-layout through per-wave LDS, then 8 WMMAs of P@V.
// ---------------------------------------------------------------------------
__global__ __launch_bounds__(128) void attn_kernel(
    const unsigned short* __restrict__ qg, const unsigned short* __restrict__ kg,
    const unsigned short* __restrict__ vg, const int* __restrict__ mask,
    const float* __restrict__ bias, unsigned short* __restrict__ outa) {
  __shared__ __align__(16) unsigned short k_lds[64 * 64];
  __shared__ __align__(16) unsigned short v_lds[64 * 64];
  __shared__ __align__(16) unsigned short p_lds[4][16 * 64];

  const int tid = threadIdx.x, lane = tid & 31, wid = tid >> 5;
  const int l16 = lane & 15, lh = lane >> 4;
  const int bh = blockIdx.y, b = bh >> 4, h = bh & 15;
  const int qm = blockIdx.x * 64 + wid * 16;

  // q fragments for K-steps d=[0,32) and d=[32,64)
  v16bf qa[2];
  const unsigned short* qp = qg + ((size_t)bh * L_ + qm + l16) * DT_ + lh * 16;
#pragma unroll
  for (int s = 0; s < 2; ++s)
#pragma unroll
    for (int g = 0; g < 2; ++g)
      __builtin_memcpy((char*)&qa[s] + 16 * g, qp + s * 32 + g * 8, 16);

  v8f acc[4] = {};
  float m_run[8], l_run[8];
#pragma unroll
  for (int r = 0; r < 8; ++r) { m_run[r] = -3.0e38f; l_run[r] = 0.f; }

  const size_t maskbase = (size_t)b * L_ * L_;
  const size_t biasbase = (size_t)h * L_ * L_;

  for (int kb = 0; kb < L_ / 64; ++kb) {
    // ---- stage K tile (contiguous 8KB) and V^T tile into LDS -------------
    const unsigned short* ksrc = kg + ((size_t)bh * L_ + kb * 64) * DT_;
#if USE_ASYNC_LDS
#pragma unroll
    for (int j = 0; j < 4; ++j) {
      int c = tid + j * 128;  // 512 chunks of 16B
      __builtin_amdgcn_global_load_async_to_lds_b128(
          (gv4_p)(ksrc + c * 8), (lv4_p)(k_lds + c * 8), 0, 0);
    }
#pragma unroll
    for (int j = 0; j < 4; ++j) {
      int c2 = tid * 4 + j;
      int dt = c2 >> 3, o8 = c2 & 7;
      const unsigned short* vsrc =
          vg + ((size_t)bh * DT_ + dt) * L_ + kb * 64 + o8 * 8;
      __builtin_amdgcn_global_load_async_to_lds_b128(
          (gv4_p)(vsrc), (lv4_p)(v_lds + dt * 64 + o8 * 8), 0, 0);
    }
#if __has_builtin(__builtin_amdgcn_s_wait_asynccnt)
    __builtin_amdgcn_s_wait_asynccnt(0);
#else
    asm volatile("s_wait_asynccnt 0" ::: "memory");
#endif
#else
#pragma unroll
    for (int j = 0; j < 4; ++j) {
      int c = tid + j * 128;
      *(uint4*)(k_lds + c * 8) = *(const uint4*)(ksrc + c * 8);
    }
#pragma unroll
    for (int j = 0; j < 4; ++j) {
      int c2 = tid * 4 + j;
      int dt = c2 >> 3, o8 = c2 & 7;
      *(uint4*)(v_lds + dt * 64 + o8 * 8) =
          *(const uint4*)(vg + ((size_t)bh * DT_ + dt) * L_ + kb * 64 + o8 * 8);
    }
#endif
    __syncthreads();

    // ---- scores: S(16x64) = q(16x64) @ K^T ------------------------------
    v8f sc[4] = {};
#pragma unroll
    for (int t = 0; t < 4; ++t)
#pragma unroll
      for (int s = 0; s < 2; ++s) {
        v16bf kf;  // B-frag: col = key (t*16+l16), K = d (s*32 + lh*16 + j)
        __builtin_memcpy(&kf, &k_lds[(t * 16 + l16) * 64 + s * 32 + lh * 16], 32);
        sc[t] = wmma_bf16(qa[s], kf, sc[t]);
      }

    // prefetch next block's bias/mask lines (streamed, zero reuse)
    if (kb + 1 < L_ / 64) {
      int prow = qm + lh * 8;
      __builtin_prefetch(bias + biasbase + (size_t)prow * L_ + (kb + 1) * 64 + lane * 2, 0, 0);
      __builtin_prefetch(mask + maskbase + (size_t)prow * L_ + (kb + 1) * 64 + lane * 2, 0, 0);
    }

    // ---- mask/bias + online softmax -------------------------------------
#pragma unroll
    for (int r = 0; r < 8; ++r) {
      const int row = qm + r + 8 * lh;
      float sr[4];
#pragma unroll
      for (int t = 0; t < 4; ++t) {
        int col = kb * 64 + t * 16 + l16;
        float s = sc[t][r] * 0.125f;  // 1/sqrt(DT)
        int   mk = __builtin_nontemporal_load(mask + maskbase + (size_t)row * L_ + col);
        float bi = __builtin_nontemporal_load(bias + biasbase + (size_t)row * L_ + col);
        sr[t] = (mk == 0 ? -10000.0f : s) + bi;
      }
      float mx = fmaxf(fmaxf(sr[0], sr[1]), fmaxf(sr[2], sr[3]));
#pragma unroll
      for (int o = 8; o >= 1; o >>= 1) mx = fmaxf(mx, __shfl_xor(mx, o, 32));
      float m_new = fmaxf(m_run[r], mx);
      float alpha = exp2f((m_run[r] - m_new) * 1.44269504f);
      float psum = 0.f;
#pragma unroll
      for (int t = 0; t < 4; ++t) {
        float p = exp2f((sr[t] - m_new) * 1.44269504f);
        psum += p;
        p_lds[wid][(r + 8 * lh) * 64 + t * 16 + l16] = f2bf_bits(p);
      }
#pragma unroll
      for (int o = 8; o >= 1; o >>= 1) psum += __shfl_xor(psum, o, 32);
      l_run[r] = l_run[r] * alpha + psum;
      m_run[r] = m_new;
#pragma unroll
      for (int t = 0; t < 4; ++t) acc[t][r] *= alpha;
    }
    asm volatile("s_wait_dscnt 0" ::: "memory");  // wave-private P transpose

    // ---- O += P(16x64) @ V(64x64) ---------------------------------------
#pragma unroll
    for (int s2 = 0; s2 < 2; ++s2) {
      v16bf pf;  // A-frag from per-wave P tile
      const unsigned short* pp = &p_lds[wid][l16 * 64 + s2 * 32 + lh * 16];
      __builtin_memcpy((char*)&pf, pp, 16);
      __builtin_memcpy((char*)&pf + 16, pp + 8, 16);
#pragma unroll
      for (int t = 0; t < 4; ++t) {
        v16bf vf;  // B-frag: col = dt (t*16+l16), K = key (s2*32 + lh*16 + j)
        __builtin_memcpy(&vf, &v_lds[(t * 16 + l16) * 64 + s2 * 32 + lh * 16], 32);
        acc[t] = wmma_bf16(pf, vf, acc[t]);
      }
    }
    __syncthreads();
  }

  // ---- normalize + store bf16 (B,L,D) for out-projection -----------------
#pragma unroll
  for (int r = 0; r < 8; ++r) {
    int row = qm + r + 8 * lh;
    float inv = 1.0f / l_run[r];
#pragma unroll
    for (int t = 0; t < 4; ++t) {
      int dt = t * 16 + l16;
      outa[((size_t)b * L_ + row) * D_ + h * DT_ + dt] = f2bf_bits(acc[t][r] * inv);
    }
  }
}

// ---------------------------------------------------------------------------
// Output projection: d_out = attn(bf16, B*L x D) @ Wo^T + bo  (f32 out)
// ---------------------------------------------------------------------------
__global__ __launch_bounds__(256) void out_proj_kernel(
    const unsigned short* __restrict__ Abf, const float* __restrict__ W,
    const float* __restrict__ bias, float* __restrict__ out) {
  const int tid = threadIdx.x, lane = tid & 31, wid = tid >> 5;
  const int l16 = lane & 15, lh = lane >> 4;
  const int m0 = blockIdx.y * 64 + (wid & 3) * 16;
  const int n0 = blockIdx.x * 64 + (wid >> 2) * 32;
  const int m_a = m0 + l16;

  v8f c0 = {0.f,0.f,0.f,0.f,0.f,0.f,0.f,0.f}, c1 = c0;
  const unsigned short* pa = Abf + (size_t)m_a * D_ + lh * 16;
  const float* pb0 = W + (size_t)(n0 + l16) * D_ + lh * 16;
  const float* pb1 = pb0 + (size_t)16 * D_;

  for (int k0 = 0; k0 < D_; k0 += 32) {
    v16bf a, b0, b1;
    __builtin_memcpy((char*)&a, pa + k0, 16);
    __builtin_memcpy((char*)&a + 16, pa + k0 + 8, 16);
#pragma unroll
    for (int q = 0; q < 4; ++q) {
      float4 f0 = *(const float4*)(pb0 + k0 + q * 4);
      float4 f1 = *(const float4*)(pb1 + k0 + q * 4);
      b0[4*q+0]=f2bf(f0.x); b0[4*q+1]=f2bf(f0.y); b0[4*q+2]=f2bf(f0.z); b0[4*q+3]=f2bf(f0.w);
      b1[4*q+0]=f2bf(f1.x); b1[4*q+1]=f2bf(f1.y); b1[4*q+2]=f2bf(f1.z); b1[4*q+3]=f2bf(f1.w);
    }
    c0 = wmma_bf16(a, b0, c0);
    c1 = wmma_bf16(a, b1, c1);
  }

#pragma unroll
  for (int r = 0; r < 8; ++r) {
    int m = m0 + r + 8 * lh;
    out[(size_t)m * D_ + n0 + l16]      = c0[r] + bias[n0 + l16];
    out[(size_t)m * D_ + n0 + 16 + l16] = c1[r] + bias[n0 + 16 + l16];
  }
}

// ---------------------------------------------------------------------------
extern "C" void kernel_launch(void* const* d_in, const int* in_sizes, int n_in,
                              void* d_out, int out_size, void* d_ws, size_t ws_size,
                              hipStream_t stream) {
  const float* Q    = (const float*)d_in[0];
  const float* K    = (const float*)d_in[1];
  const float* V    = (const float*)d_in[2];
  const int*   mask = (const int*)d_in[3];
  const float* bias = (const float*)d_in[4];
  const float* Wq = (const float*)d_in[5];  const float* bq = (const float*)d_in[6];
  const float* Wk = (const float*)d_in[7];  const float* bk = (const float*)d_in[8];
  const float* Wv = (const float*)d_in[9];  const float* bv = (const float*)d_in[10];
  const float* Wo = (const float*)d_in[11]; const float* bo = (const float*)d_in[12];

  // workspace: 4 bf16 tensors of B*L*D elements (8 MB each, 32 MB total)
  const size_t TENS = (size_t)B_ * H_ * L_ * DT_;
  unsigned short* q_ws = (unsigned short*)d_ws;
  unsigned short* k_ws = q_ws + TENS;
  unsigned short* v_ws = k_ws + TENS;   // stored transposed (B,H,DT,L)
  unsigned short* a_ws = v_ws + TENS;   // attention output (B,L,D)

  dim3 pgrid(D_ / 64, (B_ * L_) / 64);  // (16, 64)
  proj_kernel<0><<<pgrid, 256, 0, stream>>>(Q, Wq, bq, q_ws);
  proj_kernel<0><<<pgrid, 256, 0, stream>>>(K, Wk, bk, k_ws);
  proj_kernel<1><<<pgrid, 256, 0, stream>>>(V, Wv, bv, v_ws);

  attn_kernel<<<dim3(L_ / 64, B_ * H_), 128, 0, stream>>>(q_ws, k_ws, v_ws,
                                                          mask, bias, a_ws);

  out_proj_kernel<<<pgrid, 256, 0, stream>>>(a_ws, Wo, bo, (float*)d_out);
}
